// MultiHeadAttention_89713276878943
// MI455X (gfx1250) — matrix operand order repaired and checked by reading, not compiled, not run on hardware
//
#include <hip/hip_runtime.h>
#include <hip/hip_bf16.h>
#include <stdint.h>

// Problem constants (from reference)
#define B_  4
#define S_  2048
#define D_  1024
#define H_  16
#define DK_ 64
#define M_  (B_ * S_)      // 8192 rows of activations

typedef __attribute__((ext_vector_type(16))) __bf16        v16bf;
typedef __attribute__((ext_vector_type(8)))  float         v8f;
typedef __attribute__((ext_vector_type(8)))  unsigned int  v8u;
typedef __attribute__((ext_vector_type(4)))  unsigned int  u32x4;
typedef __attribute__((ext_vector_type(8)))  int           i32x8;
typedef __attribute__((ext_vector_type(4)))  int           i32x4;

// ---------------------------------------------------------------------------
// helpers
// ---------------------------------------------------------------------------
__device__ __forceinline__ unsigned short f32_to_bf16(float f) {
  __bf16 h = (__bf16)f;                       // native v_cvt on gfx1250 (RNE)
  return __builtin_bit_cast(unsigned short, h);
}

__device__ __forceinline__ v8f zero_v8f() {
  v8f z;
#pragma unroll
  for (int i = 0; i < 8; ++i) z[i] = 0.0f;
  return z;
}

// CDNA5 async copy: 16 bytes global -> LDS, tracked by ASYNCcnt (ISA §15.18).
// LDS address = low 32 bits of the flat shared-aperture pointer (ISA §10.2).
__device__ __forceinline__ void async_copy16(const void* gptr, void* lptr) {
  unsigned           loff = (unsigned)(size_t)lptr;
  unsigned long long ga   = (unsigned long long)(size_t)gptr;
  asm volatile("global_load_async_to_lds_b128 %0, %1, off"
               :: "v"(loff), "v"(ga) : "memory");
}
__device__ __forceinline__ void wait_async0() {
  asm volatile("s_wait_asynccnt 0x0" ::: "memory");
}

// CDNA5 Tensor Data Mover: DMA one 2D bf16 tile (tile_w elems x tile_h rows,
// memory row stride = stride_elems) global -> LDS, with hardware LDS padding
// of 4 DWORDs (16B) after every 16 DWORDs (64B) when pad_en (matches LDS row
// stride of 80B for a 64B data row). Descriptor bitfields per ISA §8.3/§8.4.
// Tracked by TENSORcnt; issue is per-wave (EXEC ignored).
__device__ __forceinline__ void tdm_load_tile_2d(
    const void* gaddr, void* laddr,
    unsigned tile_w_elems, unsigned tile_h,
    unsigned tensor_w_elems, unsigned tensor_h,
    unsigned stride_elems, bool pad_en) {
  unsigned long long ga = (unsigned long long)(size_t)gaddr;
  unsigned la = (unsigned)(size_t)laddr;
  u32x4 g0;
  g0[0] = 1u;                                            // count=1, user D#
  g0[1] = la;                                            // lds_addr [63:32]
  g0[2] = (unsigned)(ga & 0xFFFFFFFFu);                  // global_addr [95:64]
  g0[3] = (unsigned)((ga >> 32) & 0x01FFFFFFu) | 0x80000000u; // ga[56:32] | type=2

  i32x8 g1;
  unsigned w0 = (1u << 16);                              // data_size = 2 bytes
  if (pad_en) w0 |= (1u << 20)                           // pad_enable
             | (3u << 22)                                // pad_interval: 16 DWORDs
             | (3u << 25);                               // pad_amount:   4 DWORDs
  g1[0] = (int)w0;
  g1[1] = (int)(tensor_w_elems << 16);                   // tensor_dim0[15:0]
  g1[2] = (int)((tensor_w_elems >> 16) | (tensor_h << 16));   // dim0 hi | dim1 lo
  g1[3] = (int)((tensor_h >> 16) | (tile_w_elems << 16));     // dim1 hi | tile_dim0
  g1[4] = (int)(tile_h & 0xFFFFu);                       // tile_dim1 (tile_dim2=0)
  g1[5] = (int)stride_elems;                             // tensor_dim0_stride lo
  g1[6] = 0;                                             // stride hi | dim1_stride
  g1[7] = 0;

  i32x4 z4 = {0, 0, 0, 0};
#if __has_include(<hip/amd_detail/amd_gfx1250_TDM.h>)
  i32x8 z8 = {0, 0, 0, 0, 0, 0, 0, 0};
  __builtin_amdgcn_tensor_load_to_lds(g0, g1, z4, z4, z8, 0);
#else
  __builtin_amdgcn_tensor_load_to_lds(g0, g1, z4, z4, 0);
#endif
}

// A fragment (16x32 bf16, M x K). CDNA5 layout: lane%16 = row,
// lane/16 selects K half; VGPR v holds K pair:
//   K = (v<4 ? 0:16) + (v&3)*2 + (lane/16)*8   (+ its odd neighbor)
__device__ __forceinline__ v16bf load_a_frag(const unsigned short* lds, int ldsw,
                                             int m0, int k0, int lane) {
  const int m = lane & 15, hh = lane >> 4;
  const unsigned short* row = lds + (size_t)(m0 + m) * ldsw;
  v8u u;
#pragma unroll
  for (int v = 0; v < 8; ++v) {
    int k = k0 + ((v < 4) ? 0 : 16) + ((v & 3) * 2) + hh * 8;
    u[v] = *(const unsigned int*)(row + k);
  }
  return __builtin_bit_cast(v16bf, u);
}

// B fragment (32x16 bf16, K x N) with the "B matrix" stored row-major [N][K]
// (NT operand). CDNA5 layout: lane%16 = column n, lane/16 selects K half.
__device__ __forceinline__ v16bf load_bt_frag(const unsigned short* lds, int ldsw,
                                              int n0, int k0, int lane) {
  const int n = lane & 15, hh = lane >> 4;
  const unsigned short* row = lds + (size_t)(n0 + n) * ldsw;
  v8u u;
#pragma unroll
  for (int v = 0; v < 8; ++v) {
    int k = k0 + ((v < 4) ? 0 : 16) + ((v & 3) * 2) + hh * 8;
    u[v] = *(const unsigned int*)(row + k);
  }
  return __builtin_bit_cast(v16bf, u);
}

// reduce across the 16 lanes of a half-wave (xor 1,2,4,8 stays within a half)
__device__ __forceinline__ float half16_max(float v) {
#pragma unroll
  for (int m = 1; m <= 8; m <<= 1) v = fmaxf(v, __shfl_xor(v, m, 32));
  return v;
}
__device__ __forceinline__ float half16_sum(float v) {
#pragma unroll
  for (int m = 1; m <= 8; m <<= 1) v += __shfl_xor(v, m, 32);
  return v;
}

// ---------------------------------------------------------------------------
// kernel 0: fp32 -> bf16 cast
// ---------------------------------------------------------------------------
__global__ void cast_bf16_kernel(const float* __restrict__ src,
                                 unsigned short* __restrict__ dst, int n) {
  int i = blockIdx.x * blockDim.x + threadIdx.x;
  if (i < n) dst[i] = f32_to_bf16(src[i]);
}

// ---------------------------------------------------------------------------
// kernel 1: projection / fc GEMM:  acc = A(bf16 [M,1024]) @ W(bf16 [N,K])^T
// Block 128 thr = 4 waves, tile 64(M) x 64(N), K-step 32. Tiles staged by the
// Tensor Data Mover (one TENSOR_LOAD_TO_LDS per tile, issued by wave 0,
// double-buffered; DMA of tile t+1 overlaps WMMAs of tile t).
// ---------------------------------------------------------------------------
__device__ __forceinline__ void gemm_stage_tdm(
    const unsigned short* __restrict__ A, const unsigned short* __restrict__ W,
    unsigned short (*Asm)[40], unsigned short (*Bsm)[40],
    int m0, int n0, int kt) {
  // 64-row x 32-elem (64B) tiles; hardware pads LDS rows 64B -> 80B.
  tdm_load_tile_2d(A + (size_t)m0 * D_ + kt, &Asm[0][0], 32, 64, D_, M_, D_, true);
  tdm_load_tile_2d(W + (size_t)n0 * D_ + kt, &Bsm[0][0], 32, 64, D_, D_, D_, true);
}

template <int MODE>   // 0: head-split bf16 out, 1: V-transposed bf16 out, 2: fp32 + residual
__global__ __launch_bounds__(128) void gemm_nt_kernel(
    const unsigned short* __restrict__ A,
    const unsigned short* __restrict__ W,
    unsigned short* __restrict__ outbf,
    const float* __restrict__ res,
    float* __restrict__ outf) {
  __shared__ __align__(16) unsigned short Asm[2][64][40];
  __shared__ __align__(16) unsigned short Bsm[2][64][40];
  const int tid = threadIdx.x, lane = tid & 31, wave = tid >> 5;
  const int m0 = blockIdx.x * 64, n0 = blockIdx.y * 64;

  v8f acc[4];
#pragma unroll
  for (int nt = 0; nt < 4; ++nt) acc[nt] = zero_v8f();

  if (wave == 0) gemm_stage_tdm(A, W, Asm[0], Bsm[0], m0, n0, 0);

  for (int t = 0; t < D_ / 32; ++t) {
    if (wave == 0) __builtin_amdgcn_s_wait_tensorcnt(0);  // DMA into buf t&1 done
    __syncthreads();                                      // publish to all waves
    const int buf = t & 1;
    if (t + 1 < D_ / 32 && wave == 0)   // next tile's DMA overlaps this tile's WMMAs
      gemm_stage_tdm(A, W, Asm[buf ^ 1], Bsm[buf ^ 1], m0, n0, (t + 1) * 32);

    v16bf a = load_a_frag(&Asm[buf][0][0], 40, wave * 16, 0, lane);
#pragma unroll
    for (int nt = 0; nt < 4; ++nt) {
      v16bf b = load_bt_frag(&Bsm[buf][0][0], 40, nt * 16, 0, lane);
      acc[nt] = __builtin_amdgcn_wmma_f32_16x16x32_bf16(
          false, a, false, b, (short)0, acc[nt], false, false);
    }
  }

  const int col = lane & 15, hh = lane >> 4;
#pragma unroll
  for (int nt = 0; nt < 4; ++nt) {
#pragma unroll
    for (int r = 0; r < 8; ++r) {
      int mg = m0 + wave * 16 + r + hh * 8;
      int ng = n0 + nt * 16 + col;
      if (MODE == 0) {          // head-split [B,H,S,64]
        int bb = mg >> 11, s = mg & (S_ - 1);
        int h = ng >> 6, d = ng & 63;
        outbf[(((size_t)bb * H_ + h) * S_ + s) * DK_ + d] = f32_to_bf16(acc[nt][r]);
      } else if (MODE == 1) {   // V transposed: [B*H, 64, S]
        int bb = mg >> 11, s = mg & (S_ - 1);
        int h = ng >> 6, d = ng & 63;
        outbf[((size_t)(bb * H_ + h) * DK_ + d) * S_ + s] = f32_to_bf16(acc[nt][r]);
      } else {                  // fp32 + residual
        size_t idx = (size_t)mg * D_ + ng;
        outf[idx] = acc[nt][r] + res[idx];
      }
    }
  }
}

// ---------------------------------------------------------------------------
// kernel 2: flash attention per (b,h), 64-query block, 4 waves x 16 rows.
// K / V^T / mask tiles double-buffered via async global->LDS copies (ASYNCcnt).
// ---------------------------------------------------------------------------
__device__ __forceinline__ void attn_stage_async(
    const unsigned short* __restrict__ khead,   // [S,64] bf16
    const unsigned short* __restrict__ vThead,  // [64,S] bf16
    const unsigned char*  __restrict__ maskrow, // mask + (b*S+q0)*S
    unsigned short (*Ksm)[72], unsigned short (*Vtsm)[72],
    unsigned char (*Msm)[80], int kt, int tid) {
  // K tile: 64 rows x 64 el (128 B) = 8 chunks/row -> 512 chunks (4/thread)
  // Vt tile: same (4/thread).  Mask: 64 rows x 64 B = 256 chunks (2/thread)
#pragma unroll
  for (int i = 0; i < 4; ++i) {
    int idx = tid + i * 128;            // 0..511
    int r = idx >> 3, c = idx & 7;
    async_copy16(khead + (size_t)(kt + r) * DK_ + c * 8, &Ksm[r][c * 8]);
    async_copy16(vThead + (size_t)r * S_ + kt + c * 8,   &Vtsm[r][c * 8]);
  }
#pragma unroll
  for (int i = 0; i < 2; ++i) {
    int idx = tid + i * 128;            // 0..255
    int r = idx >> 2, c = idx & 3;
    async_copy16(maskrow + (size_t)r * S_ + kt + c * 16, &Msm[r][c * 16]);
  }
}

__global__ __launch_bounds__(128) void attn_kernel(
    const unsigned short* __restrict__ qh,    // [B*H, S, 64] bf16
    const unsigned short* __restrict__ kh,    // [B*H, S, 64] bf16
    const unsigned short* __restrict__ vT,    // [B*H, 64, S] bf16
    const unsigned char*  __restrict__ mask,  // [B, S, S] bool (1 byte)
    unsigned short* __restrict__ ctx) {       // [B, S, 1024] bf16
  __shared__ __align__(16) unsigned short Ksm[2][64][72];
  __shared__ __align__(16) unsigned short Vtsm[2][64][72];
  __shared__ __align__(16) unsigned char  Msm[2][64][80];
  __shared__ __align__(16) unsigned short Psm[64][72];

  const int tid = threadIdx.x, lane = tid & 31, wave = tid >> 5;
  const int bh = blockIdx.y, bb = bh >> 4, h = bh & 15;
  const int q0 = blockIdx.x * 64;
  const unsigned short* khead  = kh + (size_t)bh * S_ * DK_;
  const unsigned short* vThead = vT + (size_t)bh * DK_ * S_;
  const unsigned char*  mrow   = mask + ((size_t)bb * S_ + q0) * S_;
  const int col = lane & 15, hh = lane >> 4;

  // Q fragments for this wave's 16 rows (K=64 -> 2 fragments), from global
  v16bf aq[2];
  {
    const unsigned short* qrow =
        qh + (size_t)bh * S_ * DK_ + (size_t)(q0 + wave * 16 + (lane & 15)) * DK_;
#pragma unroll
    for (int f = 0; f < 2; ++f) {
      v8u u;
#pragma unroll
      for (int v = 0; v < 8; ++v) {
        int k = f * 32 + ((v < 4) ? 0 : 16) + ((v & 3) * 2) + hh * 8;
        u[v] = *(const unsigned int*)(qrow + k);
      }
      aq[f] = __builtin_bit_cast(v16bf, u);
    }
  }

  v8f o[4];
  float mrw[8], lrw[8];
#pragma unroll
  for (int nt = 0; nt < 4; ++nt) o[nt] = zero_v8f();
#pragma unroll
  for (int r = 0; r < 8; ++r) { mrw[r] = -3.0e38f; lrw[r] = 0.0f; }

  attn_stage_async(khead, vThead, mrow, Ksm[0], Vtsm[0], Msm[0], 0, tid);

  for (int t = 0; t < S_ / 64; ++t) {
    wait_async0();
    __syncthreads();
    const int buf = t & 1;
    if (t + 1 < S_ / 64)
      attn_stage_async(khead, vThead, mrow, Ksm[buf ^ 1], Vtsm[buf ^ 1],
                       Msm[buf ^ 1], (t + 1) * 64, tid);

    // S = Q K^T  (16 x 64 per wave), K-dim 64 = 2 WMMAs per n-tile
    v8f sacc[4];
#pragma unroll
    for (int nt = 0; nt < 4; ++nt) sacc[nt] = zero_v8f();
#pragma unroll
    for (int f = 0; f < 2; ++f) {
#pragma unroll
      for (int nt = 0; nt < 4; ++nt) {
        v16bf bk = load_bt_frag(&Ksm[buf][0][0], 72, nt * 16, f * 32, lane);
        sacc[nt] = __builtin_amdgcn_wmma_f32_16x16x32_bf16(
            false, aq[f], false, bk, (short)0, sacc[nt], false, false);
      }
    }

    // scale (1/sqrt(64)) + mask (mask==true -> -1e9), mask tile from LDS
#pragma unroll
    for (int nt = 0; nt < 4; ++nt) {
#pragma unroll
      for (int r = 0; r < 8; ++r) {
        float s = sacc[nt][r] * 0.125f;
        if (Msm[buf][wave * 16 + r + hh * 8][nt * 16 + col]) s = -1.0e9f;
        sacc[nt][r] = s;
      }
    }

    // online softmax update (fp32 row stats, 16-lane shuffle reductions)
#pragma unroll
    for (int r = 0; r < 8; ++r) {
      float nm = sacc[0][r];
#pragma unroll
      for (int nt = 1; nt < 4; ++nt) nm = fmaxf(nm, sacc[nt][r]);
      nm = half16_max(nm);
      float mnew = fmaxf(mrw[r], nm);
      float alpha = __expf(mrw[r] - mnew);
      float psum = 0.0f;
#pragma unroll
      for (int nt = 0; nt < 4; ++nt) {
        float p = __expf(sacc[nt][r] - mnew);
        sacc[nt][r] = p;
        psum += p;
      }
      psum = half16_sum(psum);
      lrw[r] = lrw[r] * alpha + psum;
      mrw[r] = mnew;
#pragma unroll
      for (int nt = 0; nt < 4; ++nt) o[nt][r] *= alpha;
    }

    // P -> bf16 -> LDS (wave-private rows; DS ops are in-order per wave)
#pragma unroll
    for (int nt = 0; nt < 4; ++nt) {
#pragma unroll
      for (int r = 0; r < 8; ++r) {
        Psm[wave * 16 + r + hh * 8][nt * 16 + col] = f32_to_bf16(sacc[nt][r]);
      }
    }

    // ctx += P @ V : B operand from V^T tile is contiguous (NT form)
#pragma unroll
    for (int f = 0; f < 2; ++f) {
      v16bf ap = load_a_frag(&Psm[0][0], 72, wave * 16, f * 32, lane);
#pragma unroll
      for (int nt = 0; nt < 4; ++nt) {
        v16bf bv = load_bt_frag(&Vtsm[buf][0][0], 72, nt * 16, f * 32, lane);
        o[nt] = __builtin_amdgcn_wmma_f32_16x16x32_bf16(
            false, ap, false, bv, (short)0, o[nt], false, false);
      }
    }
  }

  // finalize: divide by row sum, write [B,S,1024] with column h*64+d
#pragma unroll
  for (int nt = 0; nt < 4; ++nt) {
#pragma unroll
    for (int r = 0; r < 8; ++r) {
      int qg = q0 + wave * 16 + r + hh * 8;
      int dg = h * DK_ + nt * 16 + col;
      float val = o[nt][r] / lrw[r];
      ctx[((size_t)bb * S_ + qg) * D_ + dg] = f32_to_bf16(val);
    }
  }
}

// ---------------------------------------------------------------------------
// kernel 3: LayerNorm over rows of 1024 (weight=1, bias=0, population var)
// ---------------------------------------------------------------------------
__global__ __launch_bounds__(256) void ln_kernel(const float* __restrict__ x,
                                                 float* __restrict__ out) {
  __shared__ float ssum[8], ssq[8];
  __shared__ float stat[2];
  const int row = blockIdx.x, tid = threadIdx.x;
  const int lane = tid & 31, wid = tid >> 5;
  const float* xr = x + (size_t)row * D_;

  float s = 0.0f, s2 = 0.0f;
  for (int i = tid; i < D_; i += 256) {
    float v = xr[i];
    s += v; s2 += v * v;
  }
#pragma unroll
  for (int m = 1; m <= 16; m <<= 1) {
    s  += __shfl_xor(s,  m, 32);
    s2 += __shfl_xor(s2, m, 32);
  }
  if (lane == 0) { ssum[wid] = s; ssq[wid] = s2; }
  __syncthreads();
  if (tid == 0) {
    float ts = 0.0f, t2 = 0.0f;
#pragma unroll
    for (int i = 0; i < 8; ++i) { ts += ssum[i]; t2 += ssq[i]; }
    float mu = ts * (1.0f / D_);
    stat[0] = mu;
    stat[1] = t2 * (1.0f / D_) - mu * mu;   // population variance
  }
  __syncthreads();
  const float mu = stat[0];
  const float inv = rsqrtf(stat[1] + 1e-5f);
  for (int i = tid; i < D_; i += 256) {
    out[(size_t)row * D_ + i] = (xr[i] - mu) * inv;
  }
}

// ---------------------------------------------------------------------------
// launch
// ---------------------------------------------------------------------------
extern "C" void kernel_launch(void* const* d_in, const int* in_sizes, int n_in,
                              void* d_out, int out_size, void* d_ws, size_t ws_size,
                              hipStream_t stream) {
  (void)in_sizes; (void)n_in; (void)out_size; (void)ws_size;
  const float* Q  = (const float*)d_in[0];
  const float* K  = (const float*)d_in[1];
  const float* V  = (const float*)d_in[2];
  const unsigned char* mask = (const unsigned char*)d_in[3];  // jax bool = 1 byte
  const float* WQ = (const float*)d_in[4];
  const float* WK = (const float*)d_in[5];
  const float* WV = (const float*)d_in[6];
  const float* WF = (const float*)d_in[7];
  float* out = (float*)d_out;

  const size_t MBt = 1ull << 20;
  char* w = (char*)d_ws;
  // workspace layout (104 MB used; later stages reuse dead regions)
  unsigned short* Qbf = (unsigned short*)(w +  0 * MBt);  // 16MB, later = ctx
  unsigned short* Kbf = (unsigned short*)(w + 16 * MBt);  // 16MB, later = fc lo
  unsigned short* Vbf = (unsigned short*)(w + 32 * MBt);  // 16MB, later = fc hi
  unsigned short* WQb = (unsigned short*)(w + 48 * MBt);  // 2MB each
  unsigned short* WKb = (unsigned short*)(w + 50 * MBt);
  unsigned short* WVb = (unsigned short*)(w + 52 * MBt);
  unsigned short* WFb = (unsigned short*)(w + 54 * MBt);
  unsigned short* qh  = (unsigned short*)(w + 56 * MBt);  // 16MB head-split
  unsigned short* kh  = (unsigned short*)(w + 72 * MBt);
  unsigned short* vT  = (unsigned short*)(w + 88 * MBt);  // V^T head layout
  unsigned short* ctx = Qbf;                              // reuse (Qbf dead)
  float*          fc  = (float*)(w + 16 * MBt);           // reuse (K/Vbf dead)

  const int nAct = M_ * D_;   // 8,388,608
  const int nW   = D_ * D_;   // 1,048,576

  // 1) bf16 casts
  cast_bf16_kernel<<<(nAct + 255) / 256, 256, 0, stream>>>(Q,  Qbf, nAct);
  cast_bf16_kernel<<<(nAct + 255) / 256, 256, 0, stream>>>(K,  Kbf, nAct);
  cast_bf16_kernel<<<(nAct + 255) / 256, 256, 0, stream>>>(V,  Vbf, nAct);
  cast_bf16_kernel<<<(nW + 255) / 256,  256, 0, stream>>>(WQ, WQb, nW);
  cast_bf16_kernel<<<(nW + 255) / 256,  256, 0, stream>>>(WK, WKb, nW);
  cast_bf16_kernel<<<(nW + 255) / 256,  256, 0, stream>>>(WV, WVb, nW);
  cast_bf16_kernel<<<(nW + 255) / 256,  256, 0, stream>>>(WF, WFb, nW);

  // 2) projections -> head-split bf16 (V written transposed per head)
  dim3 gemmGrid(M_ / 64, D_ / 64), gemmBlk(128);
  gemm_nt_kernel<0><<<gemmGrid, gemmBlk, 0, stream>>>(Qbf, WQb, qh, nullptr, nullptr);
  gemm_nt_kernel<0><<<gemmGrid, gemmBlk, 0, stream>>>(Kbf, WKb, kh, nullptr, nullptr);
  gemm_nt_kernel<1><<<gemmGrid, gemmBlk, 0, stream>>>(Vbf, WVb, vT, nullptr, nullptr);

  // 3) flash attention
  attn_kernel<<<dim3(S_ / 64, B_ * H_), 128, 0, stream>>>(qh, kh, vT, mask, ctx);

  // 4) output projection + residual
  gemm_nt_kernel<2><<<gemmGrid, gemmBlk, 0, stream>>>(ctx, WFb, nullptr, Q, fc);

  // 5) LayerNorm
  ln_kernel<<<dim3(M_), 256, 0, stream>>>(fc, out);
}
